// MoERouter_81750407512754
// MI455X (gfx1250) — matrix operand order repaired and checked by reading, not compiled, not run on hardware
//
#include <hip/hip_runtime.h>

// ---------------- types ----------------
typedef __attribute__((ext_vector_type(16))) __bf16        bf16x16;
typedef __attribute__((ext_vector_type(8)))  float         f32x8;
typedef __attribute__((ext_vector_type(4)))  float         f32x4;
typedef __attribute__((ext_vector_type(4)))  unsigned int  u32x4;

struct Raw32 { u32x4 lo, hi; };   // 32 bytes, trivially copyable

// ---------------- config ----------------
constexpr int D      = 4096;      // hidden dim
constexpr int E      = 64;        // experts
constexpr int BM     = 128;       // tokens per block
constexpr int BK     = 64;        // K tile
constexpr int NT     = D / BK;    // 64 K-tiles
constexpr int XSTR   = BK + 8;    // ushort stride (144 B: b128-aligned, 36-dword stride -> conflict-free starts)
constexpr int LOGSTR = E + 3;     // 67 floats, odd stride -> conflict-free row scans

struct Tiles {
  unsigned short xt[BM][XSTR];    // bf16 raw bits, token rows
  unsigned short wt[E][XSTR];     // bf16 raw bits, expert rows (K-contiguous = column-major B)
};

union Smem {
  Tiles t[2];                     // double-buffered staging (~54 KB)
  struct {
    float logits[BM][LOGSTR];
    float probAcc[E];
    float cntAcc[E];
    float lseAcc;
  } r;
};

struct Stage { f32x4 xv[8]; f32x4 wv[4]; };   // in-flight fp32 tile data (48 VGPRs)

// pack 2 floats -> 2 bf16 (truncate) with one v_perm_b32
__device__ __forceinline__ unsigned int pack2(float f0, float f1) {
  return __builtin_amdgcn_perm(__float_as_uint(f1), __float_as_uint(f0), 0x07060302u);
}
__device__ __forceinline__ unsigned long long pack4(f32x4 v) {
  const unsigned int lo = pack2(v.x, v.y);
  const unsigned int hi = pack2(v.z, v.w);
  return ((unsigned long long)hi << 32) | lo;
}

__device__ __forceinline__ bf16x16 load_frag(const unsigned short* p0, const unsigned short* p1) {
  Raw32 raw;
  raw.lo = *reinterpret_cast<const u32x4*>(p0);   // ds_load_b128
  raw.hi = *reinterpret_cast<const u32x4*>(p1);   // ds_load_b128
  return __builtin_bit_cast(bf16x16, raw);
}

__device__ __forceinline__ void issue_loads(Stage& st, const float* __restrict__ x,
                                            const float* __restrict__ gw,
                                            int row0, int k0, int tid) {
#pragma unroll
  for (int j = 0; j < 8; ++j) {               // x tile: 128x64 fp32 = 2048 float4, 8/thread
    const int idx = tid + 256 * j;
    const int r = idx >> 4, c4 = idx & 15;
    st.xv[j] = *reinterpret_cast<const f32x4*>(x + (size_t)(row0 + r) * D + k0 + c4 * 4);
  }
#pragma unroll
  for (int j = 0; j < 4; ++j) {               // w tile: 64x64 fp32 = 1024 float4, 4/thread
    const int idx = tid + 256 * j;
    const int r = idx >> 4, c4 = idx & 15;
    st.wv[j] = *reinterpret_cast<const f32x4*>(gw + (size_t)r * D + k0 + c4 * 4);
  }
}

__device__ __forceinline__ void store_stage(const Stage& st, Tiles* buf, int tid) {
#pragma unroll
  for (int j = 0; j < 8; ++j) {
    const int idx = tid + 256 * j;
    const int r = idx >> 4, c4 = idx & 15;
    *reinterpret_cast<unsigned long long*>(&buf->xt[r][c4 * 4]) = pack4(st.xv[j]);  // ds_store_b64
  }
#pragma unroll
  for (int j = 0; j < 4; ++j) {
    const int idx = tid + 256 * j;
    const int r = idx >> 4, c4 = idx & 15;
    *reinterpret_cast<unsigned long long*>(&buf->wt[r][c4 * 4]) = pack4(st.wv[j]);
  }
}

// ------------- kernel 0: zero scratch -------------
__global__ void moe_zero(float* __restrict__ p, int n) {
  int i = blockIdx.x * blockDim.x + threadIdx.x;
  if (i < n) p[i] = 0.0f;
}

// ------------- kernel 1: GEMM + routing + partial losses -------------
__global__ __launch_bounds__(256)
void moe_router_main(const float* __restrict__ x,
                     const float* __restrict__ gw,
                     float* __restrict__ out_idx,   // [tokens*2] (indices as float)
                     float* __restrict__ out_w,     // [tokens*2]
                     float* __restrict__ gProbSum,  // [E]
                     float* __restrict__ gCnt,      // [E]
                     float* __restrict__ gLse)      // [1]
{
  __shared__ Smem smem;
  const int tid  = threadIdx.x;
  const int lane = tid & 31;
  const int wv   = tid >> 5;       // wave 0..7 -> 16-token strip
  const int hb   = lane >> 4;      // lane half (0/1), per ISA 16-bit WMMA layout
  const int l16  = lane & 15;
  const int row0 = blockIdx.x * BM;

  f32x8 acc[4] = {};               // 16x16 f32 C tiles covering experts 0..63

  Stage st;
  issue_loads(st, x, gw, row0, 0, tid);      // prologue: tile 0
  store_stage(st, &smem.t[0], tid);
  __syncthreads();

  for (int kt = 0; kt < NT; ++kt) {
    const int cur = kt & 1;
    const Tiles* tb = &smem.t[cur];

    // overlap: issue global loads for tile kt+1 while WMMAs run on tile kt
    if (kt + 1 < NT) issue_loads(st, x, gw, row0, (kt + 1) * BK, tid);
    // CDNA5 prefetch hint for tile kt+2 (global_prefetch_b8, WGP-scope: pull into near caches)
    if (kt + 2 < NT)
      __builtin_prefetch(x + (size_t)(row0 + (tid >> 1)) * D + (kt + 2) * BK + (tid & 1) * 32, 0, 3);

#pragma unroll
    for (int kk = 0; kk < BK; kk += 32) {
      // Batch all fragment loads (1 A + 4 B = 10 ds_load_b128) before any WMMA so the
      // scheduler can retire them with partial s_wait_dscnt counts under the WMMAs.
      const int arow = wv * 16 + l16;
      // A fragment 16x32: lane half selects K {h*8.., 16+h*8..} per ISA layout
      const bf16x16 a = load_frag(&tb->xt[arow][kk + hb * 8],
                                  &tb->xt[arow][kk + 16 + hb * 8]);
      bf16x16 bf[4];
#pragma unroll
      for (int t = 0; t < 4; ++t) {
        // B fragment 32x16: lane = column (expert), half selects K {h*8.., 16+h*8..}
        const int col = t * 16 + l16;
        bf[t] = load_frag(&tb->wt[col][kk + hb * 8],
                          &tb->wt[col][kk + 16 + hb * 8]);
      }
#pragma unroll
      for (int t = 0; t < 4; ++t) {
        acc[t] = __builtin_amdgcn_wmma_f32_16x16x32_bf16(
            false, a, false, bf[t], (short)0, acc[t], false, false);
      }
    }

    if (kt + 1 < NT) store_stage(st, &smem.t[cur ^ 1], tid);  // waits loads, packs via v_perm
    __syncthreads();
  }

  // spill logits to LDS (union reuse): lane holds rows M = 8*hb + r, col = t*16 + l16
#pragma unroll
  for (int t = 0; t < 4; ++t)
#pragma unroll
    for (int r = 0; r < 8; ++r)
      smem.r.logits[wv * 16 + hb * 8 + r][t * 16 + l16] = acc[t][r];
  if (tid < E) { smem.r.probAcc[tid] = 0.0f; smem.r.cntAcc[tid] = 0.0f; }
  if (tid == 255) smem.r.lseAcc = 0.0f;
  __syncthreads();

  // per-token routing: 128 threads, one token each
  if (tid < BM) {
    const float* lr = smem.r.logits[tid];
    float m1 = -INFINITY; int i1 = 0;
#pragma unroll 8
    for (int e = 0; e < E; ++e) { float v = lr[e]; if (v > m1) { m1 = v; i1 = e; } }
    float m2 = -INFINITY; int i2 = 0;
#pragma unroll 8
    for (int e = 0; e < E; ++e) { float v = lr[e]; if (e != i1 && v > m2) { m2 = v; i2 = e; } }

    float s = 0.0f;
#pragma unroll 8
    for (int e = 0; e < E; ++e) s += __expf(lr[e] - m1);
    const float inv_s = 1.0f / s;
#pragma unroll 8
    for (int e = 0; e < E; ++e) atomicAdd(&smem.r.probAcc[e], __expf(lr[e] - m1) * inv_s);

    atomicAdd(&smem.r.lseAcc, m1 + __logf(s));
    atomicAdd(&smem.r.cntAcc[i1], 1.0f);
    atomicAdd(&smem.r.cntAcc[i2], 1.0f);

    const float e2 = __expf(m2 - m1);
    const float w1 = 1.0f / (1.0f + e2);
    const int token = row0 + tid;
    out_idx[token * 2 + 0] = (float)i1;
    out_idx[token * 2 + 1] = (float)i2;
    out_w[token * 2 + 0] = w1;
    out_w[token * 2 + 1] = e2 * w1;
  }
  __syncthreads();

  if (tid < E) {
    atomicAdd(&gProbSum[tid], smem.r.probAcc[tid]);
    atomicAdd(&gCnt[tid], smem.r.cntAcc[tid]);
  }
  if (tid == 255) atomicAdd(gLse, smem.r.lseAcc);
}

// ------------- kernel 2: finalize losses -------------
__global__ void moe_router_finalize(const float* __restrict__ gProbSum,
                                    const float* __restrict__ gCnt,
                                    const float* __restrict__ gLse,
                                    float* __restrict__ out_tpe,
                                    float* __restrict__ out_bl,
                                    float* __restrict__ out_zl,
                                    int tokens)
{
  __shared__ float part[E];
  const int e = threadIdx.x;
  const float inv = 1.0f / (float)tokens;
  const float tpe = gCnt[e] * inv;
  out_tpe[e] = tpe;
  part[e] = tpe * (gProbSum[e] * inv);
  __syncthreads();
  if (e == 0) {
    float s = 0.0f;
    for (int i = 0; i < E; ++i) s += part[i];
    *out_bl = s * (float)E * 0.01f;          // AUX_LOSS_COEF
    const float ml = gLse[0] * inv;
    *out_zl = ml * ml * 0.001f;              // Z_LOSS_COEF
  }
}

// ------------- launch -------------
extern "C" void kernel_launch(void* const* d_in, const int* in_sizes, int n_in,
                              void* d_out, int out_size, void* d_ws, size_t ws_size,
                              hipStream_t stream) {
  const float* x  = (const float*)d_in[0];   // (4, 8192, 4096) fp32
  const float* gw = (const float*)d_in[1];   // (64, 4096) fp32
  const int tokens = in_sizes[0] / D;        // 32768

  float* ws       = (float*)d_ws;
  float* gProbSum = ws;                      // [64]
  float* gCnt     = ws + E;                  // [64]
  float* gLse     = ws + 2 * E;              // [1]

  float* out      = (float*)d_out;           // tuple flattened in return order
  float* out_idx  = out;                     // top_k_indices  (tokens*2)
  float* out_w    = out + tokens * 2;        // top_k_weights  (tokens*2)
  float* out_bl   = out + tokens * 4;        // balance_loss
  float* out_zl   = out + tokens * 4 + 1;    // z_loss
  float* out_tpe  = out + tokens * 4 + 2;    // tokens_per_expert (64)

  moe_zero<<<1, 256, 0, stream>>>(ws, 2 * E + 1);
  moe_router_main<<<tokens / BM, 256, 0, stream>>>(x, gw, out_idx, out_w,
                                                   gProbSum, gCnt, gLse);
  moe_router_finalize<<<1, E, 0, stream>>>(gProbSum, gCnt, gLse,
                                           out_tpe, out_bl, out_zl, tokens);
}
